// MyTransformerEncoder_71313636983125
// MI455X (gfx1250) — compile-verified, matrix operand order checked by reading, not run on hardware
//
#include <hip/hip_runtime.h>
#include <hip/hip_bf16.h>
#include <math.h>

typedef __bf16 bf16_t;
typedef __attribute__((ext_vector_type(16))) __bf16 v16bf;
typedef __attribute__((ext_vector_type(8)))  __bf16 v8bf;
typedef __attribute__((ext_vector_type(8)))  float  v8f;
typedef __attribute__((ext_vector_type(2)))  float  v2f;

#define NS 21
#define MPAD 32
#define FS 16
#define HS 256
#define EE 128
#define TS 64
#define BSZ 128
#define GN 1024          // 4*H
#define KCAT 288         // 16 + 256 + 16 pad
#define KCT 9            // KCAT/32
#define BT (BSZ*NS)      // 2688

__device__ __forceinline__ float sigm(float x) { return 1.f / (1.f + __expf(-x)); }

__device__ __forceinline__ v8f wmma_bf16(v16bf a, v16bf b, v8f c) {
    return __builtin_amdgcn_wmma_f32_16x16x32_bf16(false, a, false, b, (short)0, c, false, false);
}
__device__ __forceinline__ v8f wmma_f32(v2f a, v2f b, v8f c) {
    return __builtin_amdgcn_wmma_f32_16x16x4_f32(false, a, false, b, (short)0, c, false, false);
}

// ---- CDNA5 async global->LDS copy (ASYNCcnt path), guarded by __has_builtin ----
#if __has_builtin(__builtin_amdgcn_global_load_async_to_lds_b128)
#define HAVE_ASYNC_LDS 1
#else
#define HAVE_ASYNC_LDS 0
#endif

typedef int v4i __attribute__((vector_size(16)));
typedef __attribute__((address_space(1))) v4i as1_v4i;
typedef __attribute__((address_space(3))) v4i as3_v4i;

__device__ __forceinline__ void cp16_g2l(const bf16_t* g, bf16_t* l) {
#if HAVE_ASYNC_LDS
    __builtin_amdgcn_global_load_async_to_lds_b128((as1_v4i*)g, (as3_v4i*)l, 0, 0);
#else
    *(uint4*)l = *(const uint4*)g;
#endif
}
__device__ __forceinline__ void wait_async_lds() {
#if HAVE_ASYNC_LDS
#if __has_builtin(__builtin_amdgcn_s_wait_asynccnt)
    __builtin_amdgcn_s_wait_asynccnt(0);
#else
    asm volatile("s_wait_asynccnt 0" ::: "memory");
#endif
#endif
}

// Load a 16-element bf16 A-operand for one lane: two contiguous 8-halve segments
// (K pattern per CDNA5 16-bit A layout: {0..7,16..23} / {8..15,24..31} by lane half).
__device__ __forceinline__ v16bf ld_a(const bf16_t* p) {
    v8bf lo = *(const v8bf*)p;
    v8bf hi = *(const v8bf*)(p + 16);
    v16bf r;
#pragma unroll
    for (int i = 0; i < 8; i++) { r[i] = lo[i]; r[i + 8] = hi[i]; }
    return r;
}

// ---------------------------------------------------------------------------
// Weight pre-swizzle into per-lane WMMA B layout:
// dst[((nt*KT + kt)*32 + lane)*16 + e], lane n = nt*16 + lane%16,
// k = kt*32 + (e<8 ? (lane<16?0:8)+e : 16+(lane<16?0:8)+e-8)
// mode 0: Wcat = [W_ih(16x1024); W_hh(256x1024)]
// mode 1: transpose a[n*srcK+k]
// mode 2: kw with K reordered to [value(128) | pe(4) | pad]
// ---------------------------------------------------------------------------
__global__ void prep_kernel(const float* __restrict__ a, const float* __restrict__ b,
                            bf16_t* __restrict__ dst, int mode, int KT, int srcK,
                            int Kvalid, int total) {
    int idx = blockIdx.x * 256 + threadIdx.x;
    if (idx >= total) return;
    int e    = idx & 15;
    int lane = (idx >> 4) & 31;
    int kt   = (idx >> 9) % KT;
    int nt   = idx / (512 * KT);
    int n    = nt * 16 + (lane & 15);
    int kl   = (e < 8) ? ((lane < 16 ? 0 : 8) + e) : (16 + (lane < 16 ? 0 : 8) + (e - 8));
    int k    = kt * 32 + kl;
    float v  = 0.f;
    if (mode == 0) {
        if (k < 16)       v = a[k * 1024 + n];
        else if (k < 272) v = b[(k - 16) * 1024 + n];
    } else if (mode == 1) {
        if (k < Kvalid)   v = a[n * srcK + k];
    } else { // mode 2: key-seq K reorder
        if (k < 128)      v = a[n * srcK + 4 + k];
        else if (k < 132) v = a[n * srcK + (k - 128)];
    }
    dst[idx] = (bf16_t)v;
}

__global__ void convx_kernel(const float* __restrict__ src, bf16_t* __restrict__ dst,
                             int total) {
    int i = blockIdx.x * 256 + threadIdx.x;
    if (i < total) dst[i] = (bf16_t)src[i];
}

// ---------------------------------------------------------------------------
// h0/c0 init: h0 = G_h1 @ (x0 @ W_h1.T) + b_h1 (per batch), same for c0
// ---------------------------------------------------------------------------
__global__ __launch_bounds__(256, 1) void init_kernel(
    const float* __restrict__ x, const float* __restrict__ G_h1,
    const float* __restrict__ W_h1, const float* __restrict__ b_h1,
    const float* __restrict__ G_h2, const float* __restrict__ W_h2,
    const float* __restrict__ b_h2, float* __restrict__ h0c0) {
    __shared__ float tmp1[NS * HS];
    __shared__ float tmp2[NS * HS];
    const int b = blockIdx.x, tid = threadIdx.x;
    for (int m = 0; m < NS; m++) {
        float s1 = 0.f, s2 = 0.f;
#pragma unroll
        for (int f = 0; f < FS; f++) {
            float xv = x[((size_t)(b * TS) * NS + m) * FS + f];
            s1 += xv * W_h1[tid * FS + f];
            s2 += xv * W_h2[tid * FS + f];
        }
        tmp1[m * HS + tid] = s1;
        tmp2[m * HS + tid] = s2;
    }
    __syncthreads();
    for (int n = 0; n < NS; n++) {
        float s1 = b_h1[tid], s2 = b_h2[tid];
        for (int m = 0; m < NS; m++) {
            s1 += G_h1[n * NS + m] * tmp1[m * HS + tid];
            s2 += G_h2[n * NS + m] * tmp2[m * HS + tid];
        }
        h0c0[(size_t)(b * NS + n) * HS + tid]                   = s1;
        h0c0[(size_t)BT * HS + (size_t)(b * NS + n) * HS + tid] = s2;
    }
}

// ---------------------------------------------------------------------------
// Persistent per-batch LSTM with fused W_fc projection.
// gates = [xt|h] @ Wcat + b  -> G_lstm mix (f32 WMMA) -> activations -> h,c
// value_t = h_t @ W_fc.T + b_fc  (bf16) streamed to global.
// ---------------------------------------------------------------------------
__global__ __launch_bounds__(256, 1) void lstm_kernel(
    const bf16_t* __restrict__ xbf, const float* __restrict__ G_lstm,
    const float* __restrict__ b_lstm, const bf16_t* __restrict__ Wcat,
    const bf16_t* __restrict__ WfcT, const float* __restrict__ b_fc,
    const float* __restrict__ h0c0, bf16_t* __restrict__ value,
    float* __restrict__ outHT, float* __restrict__ outCT) {
    __shared__ __align__(16) bf16_t Abuf[MPAD * KCAT]; // 18 KB  [xt(16)|h(256)|pad(16)]
    __shared__ __align__(16) float  gates[MPAD * GN];  // 128 KB
    const int tid  = threadIdx.x;
    const int lane = tid & 31;
    const int wv   = tid >> 5;
    const int b    = blockIdx.x;
    const bool lo16 = lane < 16;
    const int ln    = lane & 15;

    for (int i = tid; i < MPAD * KCAT; i += 256) Abuf[i] = (bf16_t)0.f;
    __syncthreads();
    for (int i = tid; i < NS * HS; i += 256) {
        int n = i / HS, c = i % HS;
        Abuf[n * KCAT + 16 + c] = (bf16_t)h0c0[(size_t)(b * NS + n) * HS + c];
    }
    float creg[NS];
#pragma unroll
    for (int n = 0; n < NS; n++)
        creg[n] = h0c0[(size_t)BT * HS + (size_t)(b * NS + n) * HS + tid];

    // G_lstm (zero-padded to 32x32) in f32 WMMA A layout
    v2f Ga[2][8];
#pragma unroll
    for (int mt = 0; mt < 2; mt++)
#pragma unroll
        for (int kc = 0; kc < 8; kc++) {
            int r  = mt * 16 + ln;
            int k0 = kc * 4 + (lo16 ? 0 : 2);
            v2f g;
            g[0] = (r < NS && k0 < NS)     ? G_lstm[r * NS + k0]     : 0.f;
            g[1] = (r < NS && k0 + 1 < NS) ? G_lstm[r * NS + k0 + 1] : 0.f;
            Ga[mt][kc] = g;
        }
    __syncthreads();

#pragma unroll 1
    for (int t = 0; t < TS; t++) {
        // async-stage x_t (21 rows x 16 bf16 = 2 x 16B per row)
        if (tid < NS * 2) {
            int n = tid >> 1, hf = tid & 1;
            cp16_g2l(xbf + ((size_t)(b * TS + t) * NS + n) * FS + hf * 8,
                     &Abuf[n * KCAT + hf * 8]);
        }
        wait_async_lds();
        __syncthreads();

        // hoist A tiles for the whole step
        v16bf Areg[2][KCT];
#pragma unroll
        for (int mt = 0; mt < 2; mt++)
#pragma unroll
            for (int kc = 0; kc < KCT; kc++)
                Areg[mt][kc] = ld_a(&Abuf[(mt * 16 + ln) * KCAT + kc * 32 + (lo16 ? 0 : 8)]);

        // gate GEMM: wave w owns N-tiles 8w..8w+7 (cols 128w..128w+127)
#pragma unroll 1
        for (int ntl = 0; ntl < 8; ntl++) {
            int nt = wv * 8 + ntl;
            const bf16_t* bp = Wcat + ((size_t)nt * KCT) * 512 + lane * 16;
            __builtin_prefetch(bp + KCT * 512, 0, 3);
            v8f a0 = {}, a1 = {};
#pragma unroll
            for (int kc = 0; kc < KCT; kc++) {
                v16bf Bv = *(const v16bf*)(bp + kc * 512);
                a0 = wmma_bf16(Areg[0][kc], Bv, a0);
                a1 = wmma_bf16(Areg[1][kc], Bv, a1);
            }
            int col = nt * 16 + ln;
            float bias = b_lstm[col];
            int rb = lo16 ? 0 : 8;
#pragma unroll
            for (int j = 0; j < 8; j++) {
                gates[(j + rb) * GN + col]      = a0[j] + bias;
                gates[(16 + j + rb) * GN + col] = a1[j] + bias;
            }
        }

        // G mix in f32 WMMA, wave-local (same-wave LDS is in-order)
#pragma unroll 1
        for (int ntl = 0; ntl < 8; ntl++) {
            int nt = wv * 8 + ntl;
            int col = nt * 16 + ln;
            v8f m0 = {}, m1 = {};
#pragma unroll
            for (int kc = 0; kc < 8; kc++) {
                int mr = kc * 4 + (lo16 ? 0 : 2);
                v2f Bv;
                Bv[0] = gates[mr * GN + col];
                Bv[1] = gates[(mr + 1) * GN + col];
                m0 = wmma_f32(Ga[0][kc], Bv, m0);
                m1 = wmma_f32(Ga[1][kc], Bv, m1);
            }
            int rb = lo16 ? 0 : 8;
#pragma unroll
            for (int j = 0; j < 8; j++) {
                gates[(j + rb) * GN + col]      = m0[j];
                gates[(16 + j + rb) * GN + col] = m1[j];
            }
        }
        __syncthreads();

        // elementwise LSTM update: thread owns col=tid for all n
#pragma unroll
        for (int n = 0; n < NS; n++) {
            float iv = gates[n * GN + tid];
            float fv = gates[n * GN + 256 + tid];
            float gv = gates[n * GN + 512 + tid];
            float ov = gates[n * GN + 768 + tid];
            float c = sigm(fv) * creg[n] + sigm(iv) * tanhf(gv);
            float h = sigm(ov) * tanhf(c);
            creg[n] = c;
            Abuf[n * KCAT + 16 + tid] = (bf16_t)h;
            if (t == TS - 1) {
                outHT[(size_t)(b * NS + n) * HS + tid] = h;
                outCT[(size_t)(b * NS + n) * HS + tid] = c;
            }
        }
        __syncthreads();

        // fused value_t = h_t @ W_fc.T + b_fc : wave w owns N-tile w (16 cols of 128)
        {
            v8f a0 = {}, a1 = {};
            const bf16_t* bp = WfcT + ((size_t)wv * 8) * 512 + lane * 16;
#pragma unroll
            for (int kc = 0; kc < 8; kc++) {
                v16bf Av0 = ld_a(&Abuf[ln * KCAT + 16 + kc * 32 + (lo16 ? 0 : 8)]);
                v16bf Av1 = ld_a(&Abuf[(16 + ln) * KCAT + 16 + kc * 32 + (lo16 ? 0 : 8)]);
                v16bf Bv  = *(const v16bf*)(bp + kc * 512);
                a0 = wmma_bf16(Av0, Bv, a0);
                a1 = wmma_bf16(Av1, Bv, a1);
            }
            int col = wv * 16 + ln;
            float bias = b_fc[col];
            int rb = lo16 ? 0 : 8;
            bf16_t* vp = value + (size_t)t * BT * EE + (size_t)(b * NS) * EE + col;
#pragma unroll
            for (int j = 0; j < 8; j++) {
                int r0 = j + rb, r1 = 16 + j + rb;
                if (r0 < NS) vp[(size_t)r0 * EE] = (bf16_t)(a0[j] + bias);
                if (r1 < NS) vp[(size_t)r1 * EE] = (bf16_t)(a1[j] + bias);
            }
        }
        __syncthreads();
    }
}

// ---------------------------------------------------------------------------
// Attention (only layer 2 is live in the reference). 32 rows per workgroup.
// A-tile K layout: [value(128) | pe(4) | pad(28)] (kwT swizzled to match).
// Value tiles staged via CDNA5 async global->LDS copies.
// ---------------------------------------------------------------------------
__global__ __launch_bounds__(256, 1) void attn_kernel(
    const bf16_t* __restrict__ value, const bf16_t* __restrict__ qwT,
    const bf16_t* __restrict__ kwT, const bf16_t* __restrict__ vwT,
    const bf16_t* __restrict__ owT, const float* __restrict__ in_b,
    const float* __restrict__ ob, float* __restrict__ outp) {
    __shared__ __align__(16) bf16_t Abuf[32 * 160];   // 10 KB
    __shared__ float tile[32 * 128];                  // 16 KB
    __shared__ float qbuf[32 * 128];                  // 16 KB
    __shared__ float scores[32 * 4 * TS];             // 32 KB
    const int tid  = threadIdx.x;
    const int lane = tid & 31;
    const int wv   = tid >> 5;
    const bool lo16 = lane < 16;
    const int ln    = lane & 15;
    const int r0    = blockIdx.x * 32;
    const float scale = 0.17677669529663687f; // 1/sqrt(32)

    for (int i = tid; i < 32 * 160; i += 256) Abuf[i] = (bf16_t)0.f;
    __syncthreads();

    // q = (query @ qw.T + b) * scale ; query = value[ts-1]
#pragma unroll
    for (int u = 0; u < 2; u++) {
        int i = tid + u * 256;             // 512 chunks of 8 bf16
        int r = i >> 4, c8 = i & 15;
        cp16_g2l(value + ((size_t)(TS - 1) * BT + r0 + r) * EE + c8 * 8,
                 &Abuf[r * 160 + c8 * 8]);
    }
    wait_async_lds();
    __syncthreads();
    {
        v8f a0 = {}, a1 = {};
        const bf16_t* bp = qwT + ((size_t)wv * 4) * 512 + lane * 16;
#pragma unroll
        for (int kc = 0; kc < 4; kc++) {
            v16bf Av0 = ld_a(&Abuf[ln * 160 + kc * 32 + (lo16 ? 0 : 8)]);
            v16bf Av1 = ld_a(&Abuf[(16 + ln) * 160 + kc * 32 + (lo16 ? 0 : 8)]);
            v16bf Bv  = *(const v16bf*)(bp + kc * 512);
            a0 = wmma_bf16(Av0, Bv, a0);
            a1 = wmma_bf16(Av1, Bv, a1);
        }
        int col = wv * 16 + ln;
        float bias = in_b[2 * 384 + col];
        int rb = lo16 ? 0 : 8;
#pragma unroll
        for (int j = 0; j < 8; j++) {
            qbuf[(j + rb) * 128 + col]      = (a0[j] + bias) * scale;
            qbuf[(16 + j + rb) * 128 + col] = (a1[j] + bias) * scale;
        }
    }
    __syncthreads();

    // pass 1: k_t = [value_t | pe_t] @ kwT + b ; scores
#pragma unroll 1
    for (int t = 0; t < TS; t++) {
        float tf = (float)t;
        if (tid < 128) {
            int r = tid >> 2, pc = tid & 3;
            float pv = (pc == 0) ? sinf(tf) : (pc == 1) ? cosf(tf)
                     : (pc == 2) ? sinf(tf * 0.01f) : cosf(tf * 0.01f);
            Abuf[r * 160 + 128 + pc] = (bf16_t)pv;
        }
#pragma unroll
        for (int u = 0; u < 2; u++) {
            int i = tid + u * 256;
            int r = i >> 4, c8 = i & 15;
            cp16_g2l(value + ((size_t)t * BT + r0 + r) * EE + c8 * 8,
                     &Abuf[r * 160 + c8 * 8]);
        }
        wait_async_lds();
        __syncthreads();
        {
            v8f a0 = {}, a1 = {};
            const bf16_t* bp = kwT + ((size_t)wv * 5) * 512 + lane * 16;
#pragma unroll
            for (int kc = 0; kc < 5; kc++) {
                v16bf Av0 = ld_a(&Abuf[ln * 160 + kc * 32 + (lo16 ? 0 : 8)]);
                v16bf Av1 = ld_a(&Abuf[(16 + ln) * 160 + kc * 32 + (lo16 ? 0 : 8)]);
                v16bf Bv  = *(const v16bf*)(bp + kc * 512);
                a0 = wmma_bf16(Av0, Bv, a0);
                a1 = wmma_bf16(Av1, Bv, a1);
            }
            int col = wv * 16 + ln;
            float bias = in_b[2 * 384 + 128 + col];
            int rb = lo16 ? 0 : 8;
#pragma unroll
            for (int j = 0; j < 8; j++) {
                tile[(j + rb) * 128 + col]      = a0[j] + bias;
                tile[(16 + j + rb) * 128 + col] = a1[j] + bias;
            }
        }
        __syncthreads();
        if (tid < 128) {
            int r = tid >> 2, h = tid & 3;
            float s = 0.f;
#pragma unroll
            for (int dd = 0; dd < 32; dd++)
                s += qbuf[r * 128 + h * 32 + dd] * tile[r * 128 + h * 32 + dd];
            scores[tid * TS + t] = s;
        }
        __syncthreads();
    }

    // softmax over t
    if (tid < 128) {
        float* sp = &scores[tid * TS];
        float m = -1e30f;
        for (int t = 0; t < TS; t++) m = fmaxf(m, sp[t]);
        float s = 0.f;
        for (int t = 0; t < TS; t++) { float e = __expf(sp[t] - m); sp[t] = e; s += e; }
        float inv = 1.f / s;
        for (int t = 0; t < TS; t++) sp[t] *= inv;
    }
    __syncthreads();

    // pass 2: v_t = value_t @ vwT + b ; ctx accumulation
    float ctx[16];
#pragma unroll
    for (int e = 0; e < 16; e++) ctx[e] = 0.f;
    const int cr = tid >> 3, cc0 = (tid & 7) * 16, ch = cc0 >> 5;
#pragma unroll 1
    for (int t = 0; t < TS; t++) {
#pragma unroll
        for (int u = 0; u < 2; u++) {
            int i = tid + u * 256;
            int r = i >> 4, c8 = i & 15;
            cp16_g2l(value + ((size_t)t * BT + r0 + r) * EE + c8 * 8,
                     &Abuf[r * 160 + c8 * 8]);
        }
        wait_async_lds();
        __syncthreads();
        {
            v8f a0 = {}, a1 = {};
            const bf16_t* bp = vwT + ((size_t)wv * 4) * 512 + lane * 16;
#pragma unroll
            for (int kc = 0; kc < 4; kc++) {
                v16bf Av0 = ld_a(&Abuf[ln * 160 + kc * 32 + (lo16 ? 0 : 8)]);
                v16bf Av1 = ld_a(&Abuf[(16 + ln) * 160 + kc * 32 + (lo16 ? 0 : 8)]);
                v16bf Bv  = *(const v16bf*)(bp + kc * 512);
                a0 = wmma_bf16(Av0, Bv, a0);
                a1 = wmma_bf16(Av1, Bv, a1);
            }
            int col = wv * 16 + ln;
            float bias = in_b[2 * 384 + 256 + col];
            int rb = lo16 ? 0 : 8;
#pragma unroll
            for (int j = 0; j < 8; j++) {
                tile[(j + rb) * 128 + col]      = a0[j] + bias;
                tile[(16 + j + rb) * 128 + col] = a1[j] + bias;
            }
        }
        __syncthreads();
        float p = scores[(cr * 4 + ch) * TS + t];
#pragma unroll
        for (int e = 0; e < 16; e++) ctx[e] += p * tile[cr * 128 + cc0 + e];
        __syncthreads();
    }

    // out = query + ctx @ ow.T + ob
#pragma unroll
    for (int e = 0; e < 16; e++) Abuf[cr * 160 + cc0 + e] = (bf16_t)ctx[e];
    __syncthreads();
    {
        v8f a0 = {}, a1 = {};
        const bf16_t* bp = owT + ((size_t)wv * 4) * 512 + lane * 16;
#pragma unroll
        for (int kc = 0; kc < 4; kc++) {
            v16bf Av0 = ld_a(&Abuf[ln * 160 + kc * 32 + (lo16 ? 0 : 8)]);
            v16bf Av1 = ld_a(&Abuf[(16 + ln) * 160 + kc * 32 + (lo16 ? 0 : 8)]);
            v16bf Bv  = *(const v16bf*)(bp + kc * 512);
            a0 = wmma_bf16(Av0, Bv, a0);
            a1 = wmma_bf16(Av1, Bv, a1);
        }
        int col = wv * 16 + ln;
        float bias = ob[2 * 128 + col];
        int rb = lo16 ? 0 : 8;
#pragma unroll
        for (int j = 0; j < 8; j++) {
            int rr0 = j + rb, rr1 = 16 + j + rb;
            float q0 = (float)value[((size_t)(TS - 1) * BT + r0 + rr0) * EE + col];
            float q1 = (float)value[((size_t)(TS - 1) * BT + r0 + rr1) * EE + col];
            outp[(size_t)(r0 + rr0) * EE + col] = a0[j] + bias + q0;
            outp[(size_t)(r0 + rr1) * EE + col] = a1[j] + bias + q1;
        }
    }
}

extern "C" void kernel_launch(void* const* d_in, const int* in_sizes, int n_in,
                              void* d_out, int out_size, void* d_ws, size_t ws_size,
                              hipStream_t stream) {
    (void)in_sizes; (void)n_in; (void)out_size; (void)ws_size;
    const float* x      = (const float*)d_in[0];
    const float* G_lstm = (const float*)d_in[1];
    const float* W_ih   = (const float*)d_in[2];
    const float* W_hh   = (const float*)d_in[3];
    const float* b_lstm = (const float*)d_in[4];
    const float* G_h1   = (const float*)d_in[5];
    const float* W_h1   = (const float*)d_in[6];
    const float* b_h1   = (const float*)d_in[7];
    const float* G_h2   = (const float*)d_in[8];
    const float* W_h2   = (const float*)d_in[9];
    const float* b_h2   = (const float*)d_in[10];
    const float* W_fc   = (const float*)d_in[11];
    const float* b_fc   = (const float*)d_in[12];
    const float* qw     = (const float*)d_in[13];
    const float* kw     = (const float*)d_in[14];
    const float* vw     = (const float*)d_in[15];
    const float* in_b   = (const float*)d_in[16];
    const float* ow     = (const float*)d_in[17];
    const float* ob     = (const float*)d_in[18];

    char* ws = (char*)d_ws;
    size_t off = 0;
    bf16_t* value = (bf16_t*)(ws + off); off += (size_t)TS * BT * EE * 2;
    bf16_t* xbf   = (bf16_t*)(ws + off); off += (size_t)BSZ * TS * NS * FS * 2;
    float*  h0c0  = (float*)(ws + off);  off += (size_t)BT * HS * 2 * 4;
    bf16_t* Wcat  = (bf16_t*)(ws + off); off += (size_t)288 * 1024 * 2;
    bf16_t* WfcT  = (bf16_t*)(ws + off); off += (size_t)256 * 128 * 2;
    bf16_t* kwT   = (bf16_t*)(ws + off); off += (size_t)160 * 128 * 2;
    bf16_t* qwT   = (bf16_t*)(ws + off); off += (size_t)128 * 128 * 2;
    bf16_t* vwT   = (bf16_t*)(ws + off); off += (size_t)128 * 128 * 2;
    bf16_t* owT   = (bf16_t*)(ws + off); off += (size_t)128 * 128 * 2;

    float* outO = (float*)d_out;
    float* outH = outO + (size_t)BT * EE;
    float* outC = outH + (size_t)BT * HS;

    auto prep = [&](const float* a, const float* b, bf16_t* dst, int mode, int KT,
                    int srcK, int Kvalid, int N) {
        int total = KT * 512 * (N / 16);
        prep_kernel<<<(total + 255) / 256, 256, 0, stream>>>(a, b, dst, mode, KT, srcK,
                                                             Kvalid, total);
    };
    prep(W_ih, W_hh, Wcat, 0, 9, 0, 272, 1024);
    prep(W_fc, nullptr, WfcT, 1, 8, 256, 256, 128);
    prep(kw + 2 * 128 * 132, nullptr, kwT, 2, 5, 132, 132, 128);
    prep(qw + 2 * 128 * 128, nullptr, qwT, 1, 4, 128, 128, 128);
    prep(vw + 2 * 128 * 128, nullptr, vwT, 1, 4, 128, 128, 128);
    prep(ow + 2 * 128 * 128, nullptr, owT, 1, 4, 128, 128, 128);

    int xtot = BSZ * TS * NS * FS;
    convx_kernel<<<(xtot + 255) / 256, 256, 0, stream>>>(x, xbf, xtot);

    init_kernel<<<BSZ, 256, 0, stream>>>(x, G_h1, W_h1, b_h1, G_h2, W_h2, b_h2, h0c0);
    lstm_kernel<<<BSZ, 256, 0, stream>>>(xbf, G_lstm, b_lstm, Wcat, WfcT, b_fc, h0c0,
                                         value, outH, outC);
    attn_kernel<<<BT / 32, 256, 0, stream>>>(value, qwT, kwT, vwT, owT, in_b, ob, outO);
}
